// FastWeightMemory_31061203484858
// MI455X (gfx1250) — matrix operand order repaired and checked by reading, not compiled, not run on hardware
//
#include <hip/hip_runtime.h>
#include <hip/hip_bf16.h>

// ---------------------------------------------------------------------------
// FastWeightMemory on MI455X (gfx1250, wave32, WMMA bf16, TDM staging)
// Shapes: B=4, S=4096, H=2048, M=256, chunk=64, n_chunks=64, tokens NT=16384
// ---------------------------------------------------------------------------

#define NT 16384   // B*S tokens
#define HD 2048    // hidden
#define MD 256     // memory dim
#define CHK 64     // chunk size
#define NCH 64     // number of chunks
#define SEQ 4096   // S

typedef __attribute__((ext_vector_type(16))) __bf16 bf16x16;
typedef __attribute__((ext_vector_type(8)))  float  floatx8;

#define WMMA_BF16(a, b, c) \
  __builtin_amdgcn_wmma_f32_16x16x32_bf16(false, (a), false, (b), (short)0, (c), false, false)

union FragU {
  bf16x16 v;
  float4  f4[2];
};
union Bf8U {
  float4 f4;
  __bf16 h[8];
};

// A-fragment: 16 rows x 32 K (ISA 7.12.2, 16-bit A 16x32):
// lanes 0-15 hold K = {0..7, 16..23}; lanes 16-31 hold K = {8..15, 24..31}
__device__ __forceinline__ bf16x16 load_frag_a(const __bf16* base, int ld, int lane) {
  int r    = lane & 15;
  int koff = (lane >> 4) << 3;           // 0 or 8
  const __bf16* p = base + (size_t)r * ld + koff;
  FragU f;
  f.f4[0] = *reinterpret_cast<const float4*>(p);
  f.f4[1] = *reinterpret_cast<const float4*>(p + 16);
  return f.v;
}

// A-fragment from fp32 source, converted to bf16 on the fly
__device__ __forceinline__ bf16x16 load_frag_a_f32(const float* base, int ld, int lane) {
  int r    = lane & 15;
  int koff = (lane >> 4) << 3;
  const float* p = base + (size_t)r * ld + koff;
  float4 x0 = *reinterpret_cast<const float4*>(p);
  float4 x1 = *reinterpret_cast<const float4*>(p + 4);
  float4 x2 = *reinterpret_cast<const float4*>(p + 16);
  float4 x3 = *reinterpret_cast<const float4*>(p + 20);
  bf16x16 a;
  a[0]  = (__bf16)x0.x; a[1]  = (__bf16)x0.y; a[2]  = (__bf16)x0.z; a[3]  = (__bf16)x0.w;
  a[4]  = (__bf16)x1.x; a[5]  = (__bf16)x1.y; a[6]  = (__bf16)x1.z; a[7]  = (__bf16)x1.w;
  a[8]  = (__bf16)x2.x; a[9]  = (__bf16)x2.y; a[10] = (__bf16)x2.z; a[11] = (__bf16)x2.w;
  a[12] = (__bf16)x3.x; a[13] = (__bf16)x3.y; a[14] = (__bf16)x3.z; a[15] = (__bf16)x3.w;
  return a;
}

// B-fragment: 32 K x 16 N, from N-major ("transposed") storage srcT[n][k].
// lanes 0-15 hold K = {0..7, 16..23} of column n=lane; lanes 16-31 hold K = {8..15, 24..31}.
__device__ __forceinline__ bf16x16 load_frag_b(const __bf16* baseT, int ld, int lane) {
  int n    = lane & 15;
  int koff = (lane >> 4) << 3;           // 0 or 8
  const __bf16* p = baseT + (size_t)n * ld + koff;
  FragU f;
  f.f4[0] = *reinterpret_cast<const float4*>(p);
  f.f4[1] = *reinterpret_cast<const float4*>(p + 16);
  return f.v;
}

// ---------------------------------------------------------------------------
// Tensor Data Mover: stage a 64-row x 256-col bf16 tile (row stride 256)
// from global memory into LDS (contiguous rows). D# per CDNA5 ISA.
// Guarded: falls back to plain loads if the builtin is unavailable.
// ---------------------------------------------------------------------------
#if __has_builtin(__builtin_amdgcn_tensor_load_to_lds)
#define HAVE_TDM 1
typedef __attribute__((ext_vector_type(4))) unsigned int uint32x4;
typedef __attribute__((ext_vector_type(8))) int          int32x8;
typedef __attribute__((ext_vector_type(4))) int          int32x4;

__device__ __forceinline__ void tdm_load_tile(const __bf16* gsrc, unsigned lds_byte_addr) {
  unsigned long long ga = (unsigned long long)(uintptr_t)gsrc;
  uint32x4 g0;
  g0[0] = 1u;                                              // count=1 (valid D#)
  g0[1] = lds_byte_addr;                                   // lds_addr[31:0]
  g0[2] = (unsigned)(ga & 0xFFFFFFFFu);                    // global_addr lo
  g0[3] = ((unsigned)(ga >> 32) & 0x01FFFFFFu) | (2u << 30); // addr hi | type=2
  int32x8 g1;
  g1[0] = (int)(1u << 16);        // data_size=1 (2 bytes), mask=0, no pad
  g1[1] = (int)(256u << 16);      // tensor_dim0[15:0]=256  (bits 79:48)
  g1[2] = (int)(64u << 16);       // tensor_dim0 hi=0 | tensor_dim1=64
  g1[3] = (int)(256u << 16);      // tensor_dim1 hi=0 | tile_dim0=256
  g1[4] = 64;                     // tile_dim1=64, tile_dim2=0
  g1[5] = 256;                    // tensor_dim0_stride=256 (elements)
  g1[6] = 0;
  g1[7] = 0;
  int32x4 z4 = {0, 0, 0, 0};
#if __clang_major__ >= 23
  int32x8 z8 = {0, 0, 0, 0, 0, 0, 0, 0};
  __builtin_amdgcn_tensor_load_to_lds(g0, g1, z4, z4, z8, 0);
#else
  __builtin_amdgcn_tensor_load_to_lds(g0, g1, z4, z4, 0);
#endif
}
#endif

// ---------------------------------------------------------------------------
// 1) fp32 -> bf16 convert (weights only; x is converted on the fly)
// ---------------------------------------------------------------------------
__global__ void k_cvt_bf16(const float* __restrict__ in, __bf16* __restrict__ out, int n) {
  int i = blockIdx.x * blockDim.x + threadIdx.x;
  int stride = gridDim.x * blockDim.x;
  for (; i < n; i += stride) out[i] = (__bf16)in[i];
}

// ---------------------------------------------------------------------------
// 2) Fused projection: q/k/v = x @ W{q,k,v}^T with on-the-fly cvt of x,
//    in-block row L2-normalization of k,v, bf16 outputs.
//    Block = 256 threads (8 waves), 32 rows x 256 cols, all 3 weights.
// ---------------------------------------------------------------------------
__global__ __launch_bounds__(256) void k_proj_fused(
    const float* __restrict__ x,
    const __bf16* __restrict__ wqb, const __bf16* __restrict__ wkb,
    const __bf16* __restrict__ wvb,
    __bf16* __restrict__ qbf, __bf16* __restrict__ kbf, __bf16* __restrict__ vbf) {
  const int tid  = threadIdx.x;
  const int wid  = tid >> 5;
  const int lane = tid & 31;
  const int rloc = wid >> 2;
  const int cg   = wid & 3;
  const int rsel = (lane >> 4) << 3;
  const int cl   = lane & 15;
  const int row0 = blockIdx.x * 32;

  floatx8 acc[3][4];
#pragma unroll
  for (int w = 0; w < 3; ++w)
#pragma unroll
    for (int j = 0; j < 4; ++j) acc[w][j] = (floatx8){0, 0, 0, 0, 0, 0, 0, 0};

  const __bf16* Wp[3] = {wqb, wkb, wvb};
  const float* abase = x + (size_t)(row0 + rloc * 16) * HD;

  for (int ks = 0; ks < HD; ks += 32) {
    __builtin_prefetch(abase + ks + 256, 0, 0);           // global_prefetch_b8
    bf16x16 a = load_frag_a_f32(abase + ks, HD, lane);
#pragma unroll
    for (int w = 0; w < 3; ++w) {
      bf16x16 bfr[4];
#pragma unroll
      for (int j = 0; j < 4; ++j)
        bfr[j] = load_frag_b(Wp[w] + (size_t)(cg * 64 + j * 16) * HD + ks, HD, lane);
#pragma unroll
      for (int j = 0; j < 4; ++j) acc[w][j] = WMMA_BF16(a, bfr[j], acc[w][j]);
    }
  }

#define PSTR 264
  __shared__ float tile[32 * PSTR];
  __shared__ float part[32 * 8];
  __shared__ float rinv[32];
  __bf16* Op[3] = {qbf, kbf, vbf};

  const int trow = tid >> 3;
  const int tseg = tid & 7;

  for (int w = 0; w < 3; ++w) {
#pragma unroll
    for (int j = 0; j < 4; ++j) {
      int col = cg * 64 + j * 16 + cl;
#pragma unroll
      for (int i = 0; i < 8; ++i)
        tile[(rloc * 16 + i + rsel) * PSTR + col] = acc[w][j][i];
    }
    __syncthreads();
    float ps = 0.0f;
#pragma unroll
    for (int c = 0; c < 32; ++c) {
      float v = tile[trow * PSTR + tseg * 32 + c];
      ps += v * v;
    }
    part[trow * 8 + tseg] = ps;
    __syncthreads();
    if (tid < 32) {
      float s = 0.0f;
#pragma unroll
      for (int c = 0; c < 8; ++c) s += part[tid * 8 + c];
      rinv[tid] = (w == 0) ? 1.0f : (1.0f / fmaxf(sqrtf(s), 1e-12f));
    }
    __syncthreads();
    float inv = rinv[trow];
    __bf16* o = Op[w] + (size_t)(row0 + trow) * MD + tseg * 32;
#pragma unroll
    for (int c = 0; c < 32; ++c) o[c] = (__bf16)(tile[trow * PSTR + tseg * 32 + c] * inv);
    __syncthreads();
  }
}

// ---------------------------------------------------------------------------
// 3) Persistent sequential scan: one workgroup, 32 waves.
//    M (256x256 fp32) lives in WMMA accumulators (8 x v8f per wave).
//    LDS: region A (256x264 bf16) = M mirror, then v^T chunk
//         region B (256x264 bf16) = k^T chunk ; 64 fp32 clip slots
//    Chunk staging via TDM (tensor_load_to_lds + s_wait_tensorcnt) when
//    available, else direct vector loads; then an in-LDS register transpose.
// ---------------------------------------------------------------------------
#define LSTR 264
#define SCAN_LDS_BYTES (2 * 256 * LSTR * 2 + 256)

__global__ __launch_bounds__(1024) void k_scan(const __bf16* __restrict__ qb,
                                               const __bf16* __restrict__ kb,
                                               const __bf16* __restrict__ vb,
                                               __bf16* __restrict__ rb) {
  extern __shared__ char smem[];
  __bf16* ldsA  = (__bf16*)smem;                        // M mirror / v raw / v^T
  __bf16* ldsB  = (__bf16*)(smem + 256 * LSTR * 2);     // k raw / k^T
  float*  slots = (float*)(smem + 2 * 256 * LSTR * 2);  // per-chunk sumsq

  const int tid  = threadIdx.x;
  const int wid  = tid >> 5;
  const int lane = tid & 31;
  const int ti   = wid >> 1;
  const int cg   = wid & 1;
  const int rsel = (lane >> 4) << 3;
  const int cl   = lane & 15;

  const float decay_pow = 0.525595909510321f;   // 0.99^64
  const float invden    = 1.0f / 256.0f;        // 1/(B*chunk)

  floatx8 acc[8];
#pragma unroll
  for (int j = 0; j < 8; ++j) acc[j] = (floatx8){0, 0, 0, 0, 0, 0, 0, 0};
  if (tid < NCH) slots[tid] = 0.0f;
  __syncthreads();

  const int bb  = ti >> 2;
  const int inb = (ti * 16) & 63;

  for (int ch = 0; ch < NCH; ++ch) {
    // --- 1. mirror fp32 M into LDS bf16 (row-major [n][m]) ------------------
#pragma unroll
    for (int j = 0; j < 8; ++j) {
      int m = (cg * 8 + j) * 16 + cl;
#pragma unroll
      for (int i = 0; i < 8; ++i)
        ldsA[(ti * 16 + i + rsel) * LSTR + m] = (__bf16)acc[j][i];
    }
    __syncthreads();

    // --- 2. read phase: r = q_chunk @ M^T (BEFORE write) --------------------
    const int grow0 = bb * SEQ + ch * CHK + inb;
    const __bf16* qbase = qb + (size_t)grow0 * MD;
#pragma unroll
    for (int jg = 0; jg < 2; ++jg) {
      floatx8 rc[4];
#pragma unroll
      for (int j = 0; j < 4; ++j) rc[j] = (floatx8){0, 0, 0, 0, 0, 0, 0, 0};
#pragma unroll
      for (int ks = 0; ks < MD; ks += 32) {
        bf16x16 a = load_frag_a(qbase + ks, MD, lane);
        bf16x16 bfr[4];
#pragma unroll
        for (int j = 0; j < 4; ++j) {
          int nj = (cg * 8 + jg * 4 + j) * 16;
          bfr[j] = load_frag_b(ldsA + nj * LSTR + ks, LSTR, lane);
        }
#pragma unroll
        for (int j = 0; j < 4; ++j) rc[j] = WMMA_BF16(a, bfr[j], rc[j]);
      }
#pragma unroll
      for (int j = 0; j < 4; ++j) {
        int nj = (cg * 8 + jg * 4 + j) * 16;
#pragma unroll
        for (int i = 0; i < 8; ++i)
          rb[(size_t)(grow0 + i + rsel) * MD + nj + cl] = (__bf16)rc[j][i];
      }
    }
    __syncthreads();

#ifdef HAVE_TDM
    // --- 3a. TDM: DMA raw k rows -> ldsB, raw v rows -> ldsA (stride 256) ---
    if (tid == 0) {
#pragma unroll
      for (int b = 0; b < 4; ++b) {
        size_t goff = (size_t)(b * SEQ + ch * CHK) * MD;
        unsigned lds_off = (unsigned)(b * CHK * MD * 2);
        tdm_load_tile(kb + goff, (unsigned)(uintptr_t)(void*)ldsB + lds_off);
        tdm_load_tile(vb + goff, (unsigned)(uintptr_t)(void*)ldsA + lds_off);
      }
#if __has_builtin(__builtin_amdgcn_s_wait_tensorcnt)
      __builtin_amdgcn_s_wait_tensorcnt(0);
#endif
    }
    __syncthreads();
    // --- 3b. in-LDS transpose: raw [kr][m] (ld 256) -> [m][kr] (ld LSTR) ----
    {
      float4 kbuf[8], vbuf[8];
#pragma unroll
      for (int it = 0; it < 8; ++it) {
        int idx = it * 1024 + tid;
        int sg = idx & 31, kr = idx >> 5;
        kbuf[it] = *reinterpret_cast<const float4*>(ldsB + kr * MD + sg * 8);
        vbuf[it] = *reinterpret_cast<const float4*>(ldsA + kr * MD + sg * 8);
      }
      __syncthreads();
#pragma unroll
      for (int it = 0; it < 8; ++it) {
        int idx = it * 1024 + tid;
        int sg = idx & 31, kr = idx >> 5;
        Bf8U ku, vu;
        ku.f4 = kbuf[it];
        vu.f4 = vbuf[it];
#pragma unroll
        for (int e = 0; e < 8; ++e) {
          int m = sg * 8 + e;
          ldsB[m * LSTR + kr] = ku.h[e];
          ldsA[m * LSTR + kr] = (__bf16)((float)vu.h[e] * invden);
        }
      }
    }
#else
    // --- 3. direct staged transposed load ----------------------------------
    for (int it = 0; it < 8; ++it) {
      int idx = it * 1024 + tid;
      int seg = idx & 31;
      int kr  = idx >> 5;
      int gr  = (kr >> 6) * SEQ + ch * CHK + (kr & 63);
      Bf8U kv, vv;
      kv.f4 = *reinterpret_cast<const float4*>(kb + (size_t)gr * MD + seg * 8);
      vv.f4 = *reinterpret_cast<const float4*>(vb + (size_t)gr * MD + seg * 8);
#pragma unroll
      for (int e = 0; e < 8; ++e) {
        int m = seg * 8 + e;
        ldsB[m * LSTR + kr] = kv.h[e];
        ldsA[m * LSTR + kr] = (__bf16)((float)vv.h[e] * invden);
      }
    }
#endif
    __syncthreads();

    // --- 4. write phase: M = decay*M + v^T k / denom ------------------------
#pragma unroll
    for (int j = 0; j < 8; ++j)
#pragma unroll
      for (int i = 0; i < 8; ++i) acc[j][i] *= decay_pow;

    for (int ks = 0; ks < MD; ks += 32) {
      bf16x16 a = load_frag_a(ldsA + (ti * 16) * LSTR + ks, LSTR, lane);
#pragma unroll
      for (int jg = 0; jg < 2; ++jg) {
        bf16x16 bfr[4];
#pragma unroll
        for (int j = 0; j < 4; ++j)
          bfr[j] = load_frag_b(ldsB + ((cg * 8 + jg * 4 + j) * 16) * LSTR + ks, LSTR, lane);
#pragma unroll
        for (int j = 0; j < 4; ++j)
          acc[jg * 4 + j] = WMMA_BF16(a, bfr[j], acc[jg * 4 + j]);
      }
    }

    // --- 5. Frobenius-norm clip --------------------------------------------
    float s = 0.0f;
#pragma unroll
    for (int j = 0; j < 8; ++j)
#pragma unroll
      for (int i = 0; i < 8; ++i) s += acc[j][i] * acc[j][i];
#pragma unroll
    for (int off = 16; off; off >>= 1) s += __shfl_xor(s, off, 32);
    if (lane == 0) atomicAdd(&slots[ch], s);      // ds_add_f32
    __syncthreads();
    float nrm   = sqrtf(slots[ch]);
    float scale = fminf(1.0f, 10.0f / fmaxf(nrm, 1e-30f));
#pragma unroll
    for (int j = 0; j < 8; ++j)
#pragma unroll
      for (int i = 0; i < 8; ++i) acc[j][i] *= scale;
  }
}

// ---------------------------------------------------------------------------
// 4) Fused output projection + per-row norm clip.
//    Block = 1024 threads (32 waves) owns a full 16 x 2048 output row-block,
//    staged in ~128KB LDS, clipped, stored once.
// ---------------------------------------------------------------------------
#define OSTR 2056
#define OUT_LDS_BYTES (16 * OSTR * 4 + 16 * 64 * 4 + 64)

__global__ __launch_bounds__(1024) void k_outproj_fused(
    const __bf16* __restrict__ rbf, const __bf16* __restrict__ wob,
    float* __restrict__ out) {
  extern __shared__ char smem[];
  float* tile = (float*)smem;
  float* part = (float*)(smem + 16 * OSTR * 4);
  float* rscl = (float*)(smem + 16 * OSTR * 4 + 16 * 64 * 4);

  const int tid  = threadIdx.x;
  const int wid  = tid >> 5;
  const int lane = tid & 31;
  const int rsel = (lane >> 4) << 3;
  const int cl   = lane & 15;
  const int row0 = blockIdx.x * 16;

  floatx8 acc[4];
#pragma unroll
  for (int j = 0; j < 4; ++j) acc[j] = (floatx8){0, 0, 0, 0, 0, 0, 0, 0};

  const __bf16* abase = rbf + (size_t)row0 * MD;
#pragma unroll
  for (int ks = 0; ks < MD; ks += 32) {
    bf16x16 a = load_frag_a(abase + ks, MD, lane);
    bf16x16 bfr[4];
#pragma unroll
    for (int j = 0; j < 4; ++j)
      bfr[j] = load_frag_b(wob + (size_t)(wid * 64 + j * 16) * MD + ks, MD, lane);
#pragma unroll
    for (int j = 0; j < 4; ++j) acc[j] = WMMA_BF16(a, bfr[j], acc[j]);
  }

#pragma unroll
  for (int j = 0; j < 4; ++j) {
    int col = wid * 64 + j * 16 + cl;
#pragma unroll
    for (int i = 0; i < 8; ++i) tile[(i + rsel) * OSTR + col] = acc[j][i];
  }
  __syncthreads();

  const int trow = tid >> 6;
  const int tseg = tid & 63;
  float ps = 0.0f;
#pragma unroll
  for (int c = 0; c < 32; ++c) {
    float v = tile[trow * OSTR + tseg * 32 + c];
    ps += v * v;
  }
  part[trow * 64 + tseg] = ps;
  __syncthreads();
  if (tid < 16) {
    float s = 0.0f;
    for (int c = 0; c < 64; ++c) s += part[tid * 64 + c];
    float nrm = sqrtf(s);
    rscl[tid] = fminf(1.0f, 10.0f / fmaxf(nrm, 1e-6f));
  }
  __syncthreads();
  float scale = rscl[trow];
  float* o = out + (size_t)(row0 + trow) * HD + tseg * 32;
#pragma unroll
  for (int c = 0; c < 32; ++c) o[c] = tile[trow * OSTR + tseg * 32 + c] * scale;
}

// ---------------------------------------------------------------------------
// Host launcher
// ---------------------------------------------------------------------------
extern "C" void kernel_launch(void* const* d_in, const int* in_sizes, int n_in,
                              void* d_out, int out_size, void* d_ws, size_t ws_size,
                              hipStream_t stream) {
  (void)in_sizes; (void)n_in; (void)out_size; (void)ws_size;
  const float* x  = (const float*)d_in[0];
  const float* Wq = (const float*)d_in[1];
  const float* Wk = (const float*)d_in[2];
  const float* Wv = (const float*)d_in[3];
  const float* Wo = (const float*)d_in[4];
  // d_in[5] = M0 (zeros, matches register init); d_in[6] = chunk_size (=64)
  float* out = (float*)d_out;

  char* ws = (char*)d_ws;
  size_t off = 0;
  auto carve = [&](size_t bytes) -> void* {
    void* p = ws + off;
    off += (bytes + 255) & ~(size_t)255;
    return p;
  };
  __bf16* wqbf = (__bf16*)carve((size_t)MD * HD * 2);
  __bf16* wkbf = (__bf16*)carve((size_t)MD * HD * 2);
  __bf16* wvbf = (__bf16*)carve((size_t)MD * HD * 2);
  __bf16* wobf = (__bf16*)carve((size_t)HD * MD * 2);
  __bf16* qbf  = (__bf16*)carve((size_t)NT * MD * 2);
  __bf16* kbf  = (__bf16*)carve((size_t)NT * MD * 2);
  __bf16* vbf  = (__bf16*)carve((size_t)NT * MD * 2);
  __bf16* rbf  = (__bf16*)carve((size_t)NT * MD * 2);

  // 1) convert weights to bf16
  k_cvt_bf16<<<512, 256, 0, stream>>>(Wq, wqbf, MD * HD);
  k_cvt_bf16<<<512, 256, 0, stream>>>(Wk, wkbf, MD * HD);
  k_cvt_bf16<<<512, 256, 0, stream>>>(Wv, wvbf, MD * HD);
  k_cvt_bf16<<<512, 256, 0, stream>>>(Wo, wobf, HD * MD);

  // 2) fused q/k/v projection + normalization (x fp32 read exactly once)
  k_proj_fused<<<NT / 32, 256, 0, stream>>>(x, wqbf, wkbf, wvbf, qbf, kbf, vbf);

  // 3) sequential fast-weight scan (single persistent workgroup, TDM staging)
  k_scan<<<1, 1024, SCAN_LDS_BYTES, stream>>>(qbf, kbf, vbf, rbf);

  // 4) fused output projection + row norm clip (d_out written exactly once)
  k_outproj_fused<<<NT / 16, 1024, OUT_LDS_BYTES, stream>>>(rbf, wobf, out);
}
